// Meg_InteractionBlock_51788715655492
// MI455X (gfx1250) — matrix operand order repaired and checked by reading, not compiled, not run on hardware
//
#include <hip/hip_runtime.h>

// ---------------------------------------------------------------------------
// SchNet interaction block, fused for MI455X (gfx1250, wave32, WMMA bf16).
//   K0 prep : fp32 weights -> bf16 (w1 padded K 50->64), row-major W^T columns
//   K1 zero : agg = 0
//   K2 gemm : xh = x @ lin1^T                     (WMMA bf16, f32 accum)
//   K3 edge : h = ssp(ea@w1^T+b1); W=(h@w2^T+b2)*C; atomicAdd(agg[dst], xh[src]*W)
//   K4 gemm : out = ssp(agg @ lin2^T + b)         (WMMA bf16, f32 accum)
// Weight tiles are staged HBM/L2 -> LDS via gfx1250 async-to-LDS (ASYNCcnt).
// ---------------------------------------------------------------------------

typedef __attribute__((ext_vector_type(16))) __bf16 v16bf;
typedef __attribute__((ext_vector_type(8)))  __bf16 v8bf;
typedef __attribute__((ext_vector_type(8)))  float  v8f;

#define HDIM   128      // hidden / filter width
#define GDIM   50       // gaussian features
#define GPAD   64       // padded K for GEMM1
#define WSTR   136      // LDS row stride in bf16 elems (128 + 8 pad -> bank rotate)
#define EWAVES 6        // edge-tiles (waves) per block

__device__ __forceinline__ float sspf(float x) {
  // shifted softplus, numerically stable
  float t = __logf(1.0f + __expf(-fabsf(x))) + fmaxf(x, 0.0f);
  return t - 0.69314718055994530942f;
}
__device__ __forceinline__ float bcast_f(float v, int src_lane) {
  return __int_as_float(__builtin_amdgcn_ds_bpermute(src_lane << 2, __float_as_int(v)));
}
__device__ __forceinline__ int bcast_i(int v, int src_lane) {
  return __builtin_amdgcn_ds_bpermute(src_lane << 2, v);
}
__device__ __forceinline__ v16bf pack16(v8bf lo, v8bf hi) {
  v16bf r;
#pragma unroll
  for (int i = 0; i < 8; i++) { r[i] = lo[i]; r[i + 8] = hi[i]; }
  return r;
}

// gfx1250 async copy: 16B from (gbase + goff) directly into LDS, no VGPR round
// trip. GVS mode: SADDR = 64-bit SGPR base, VADDR = 32-bit byte offset.
// Tracked by ASYNCcnt; wait with s_wait_asynccnt before the barrier.
__device__ __forceinline__ void async_copy_b128(const void* gbase, unsigned goff,
                                                void* lds_ptr) {
  unsigned lds = (unsigned)(size_t)lds_ptr;  // low 32 bits = LDS offset
  asm volatile("global_load_async_to_lds_b128 %0, %1, %2"
               :: "v"(lds), "v"(goff), "s"(gbase) : "memory");
}
__device__ __forceinline__ void wait_async0() {
  asm volatile("s_wait_asynccnt 0" ::: "memory");
}

// ---------------------------------------------------------------------------
__global__ void prep_weights_kernel(const float* __restrict__ w1, const float* __restrict__ w2,
                                    const float* __restrict__ l1, const float* __restrict__ l2,
                                    __bf16* __restrict__ w1b, __bf16* __restrict__ w2b,
                                    __bf16* __restrict__ l1b, __bf16* __restrict__ l2b) {
  int i = blockIdx.x * blockDim.x + threadIdx.x;
  if (i < HDIM * GPAD) {
    int f = i >> 6, g = i & 63;
    w1b[i] = (__bf16)((g < GDIM) ? w1[f * GDIM + g] : 0.0f);
  }
  if (i < HDIM * HDIM) {
    w2b[i] = (__bf16)w2[i];
    l1b[i] = (__bf16)l1[i];
    l2b[i] = (__bf16)l2[i];
  }
}

__global__ void zero_kernel(float* __restrict__ p, long n) {
  long i = (long)blockIdx.x * blockDim.x + threadIdx.x;
  if (i < n) p[i] = 0.0f;
}

// ---------------------------------------------------------------------------
// Row-tile GEMM: out[r, :] = (ssp?)(A[r, :128] @ Wb^T (+ bias))
// One wave per 16-row tile. Wb stored so that B-matrix column n = Wb row n.
__global__ __launch_bounds__(256) void gemm_rowtile_kernel(
    const float* __restrict__ A, const __bf16* __restrict__ Wb,
    const float* __restrict__ bias, float* __restrict__ out,
    int rows, int apply_ssp) {
  __shared__ __bf16 s_w[HDIM * WSTR];
  for (int i = threadIdx.x; i < HDIM * 16; i += blockDim.x) {
    int r = i >> 4, c8 = (i & 15) << 3;
    async_copy_b128(Wb, (unsigned)((r * HDIM + c8) * sizeof(__bf16)),
                    s_w + r * WSTR + c8);
  }
  wait_async0();
  __syncthreads();

  const int wave = threadIdx.x >> 5, lane = threadIdx.x & 31;
  const int hi = lane >> 4, l16 = lane & 15;
  long tile = (long)blockIdx.x * (blockDim.x >> 5) + wave;
  long row0 = tile * 16;
  if (row0 >= rows) return;

  // A fragments: 16x32 bf16 per k-step; lane<16 holds row l16 K {0..15}+32ks (lane>=16: 16..31)
  long arow = row0 + l16;
  bool rv = (arow < rows);
  const float* ap = A + (rv ? arow : 0) * (long)HDIM;
  v16bf a[4];
#pragma unroll
  for (int ks = 0; ks < 4; ks++) {
#pragma unroll
    for (int c = 0; c < 2; c++) {
      int kk = ks * 32 + (hi ? 16 : 0) + c * 8;
      float4 f0 = make_float4(0.f, 0.f, 0.f, 0.f);
      float4 f1 = make_float4(0.f, 0.f, 0.f, 0.f);
      if (rv) { f0 = *(const float4*)(ap + kk); f1 = *(const float4*)(ap + kk + 4); }
      int lo = c * 8;
      a[ks][lo + 0] = (__bf16)f0.x; a[ks][lo + 1] = (__bf16)f0.y;
      a[ks][lo + 2] = (__bf16)f0.z; a[ks][lo + 3] = (__bf16)f0.w;
      a[ks][lo + 4] = (__bf16)f1.x; a[ks][lo + 5] = (__bf16)f1.y;
      a[ks][lo + 6] = (__bf16)f1.z; a[ks][lo + 7] = (__bf16)f1.w;
    }
  }

  for (int nt = 0; nt < 8; nt++) {
    int n = nt * 16 + l16;
    float binit = bias ? bias[n] : 0.0f;
    v8f acc;
#pragma unroll
    for (int i = 0; i < 8; i++) acc[i] = binit;
#pragma unroll
    for (int ks = 0; ks < 4; ks++) {
      int koff = ks * 32 + (hi ? 16 : 0);
      v8bf b0 = *(const v8bf*)(s_w + n * WSTR + koff);
      v8bf b1 = *(const v8bf*)(s_w + n * WSTR + koff + 8);
      acc = __builtin_amdgcn_wmma_f32_16x16x32_bf16(
          false, a[ks], false, pack16(b0, b1), (short)0, acc, false, false);
    }
#pragma unroll
    for (int j = 0; j < 8; j++) {
      long r = row0 + j + 8 * hi;   // C/D layout: VGPR j -> M = j + 8*hi
      if (r < rows) {
        float v = acc[j];
        if (apply_ssp) v = sspf(v);
        out[r * HDIM + n] = v;
      }
    }
  }
}

// ---------------------------------------------------------------------------
// Fused edge kernel: one wave per 16-edge tile, EWAVES waves per block.
__global__ __launch_bounds__(EWAVES * 32) void edge_kernel(
    const float* __restrict__ edge_attr,    // [E, 50]
    const float* __restrict__ edge_weight,  // [E]
    const long long* __restrict__ eidx,     // [2, E] int64
    const float* __restrict__ b1, const float* __restrict__ b2,
    const __bf16* __restrict__ w1b,         // [128, 64]  (W1^T columns = rows)
    const __bf16* __restrict__ w2b,         // [128, 128] (W2^T columns = rows)
    const float* __restrict__ xh,           // [N, 128]
    float* __restrict__ agg,                // [N, 128]
    int nE) {
  __shared__ __bf16 s_w2[HDIM * WSTR];
  __shared__ __bf16 s_h[EWAVES][16 * WSTR];
  for (int i = threadIdx.x; i < HDIM * 16; i += blockDim.x) {
    int r = i >> 4, c8 = (i & 15) << 3;
    async_copy_b128(w2b, (unsigned)((r * HDIM + c8) * sizeof(__bf16)),
                    s_w2 + r * WSTR + c8);
  }
  wait_async0();
  __syncthreads();

  const int wave = threadIdx.x >> 5, lane = threadIdx.x & 31;
  const int hi = lane >> 4, l16 = lane & 15;
  long e0 = ((long)blockIdx.x * EWAVES + wave) * 16;
  if (e0 >= nE) return;

  // per-edge scalars (lanes 0-15 own edges; duplicated on upper half)
  long e = e0 + l16;
  float Cv = 0.0f; int sv = 0, dv = -1;
  bool ev = (e < nE);
  if (ev) {
    Cv = 0.5f * (__cosf(edge_weight[e] * 0.31415926535897932f) + 1.0f);
    sv = (int)eidx[e];
    dv = (int)eidx[(long)nE + e];
  }
  float Cm[8]; int sm[8], dm[8];
#pragma unroll
  for (int j = 0; j < 8; j++) {
    int mm = j + 8 * hi;
    Cm[j] = bcast_f(Cv, mm);
    sm[j] = bcast_i(sv, mm);
    dm[j] = bcast_i(dv, mm);
  }

  // --- GEMM1 A fragments from edge_attr (K padded 50 -> 64, zeros past 50)
  const float* ap = edge_attr + (ev ? e : 0) * (long)GDIM;
  v16bf a1[2];
#pragma unroll
  for (int ks = 0; ks < 2; ks++) {
#pragma unroll
    for (int c = 0; c < 2; c++) {
      int g0 = ks * 32 + (hi ? 16 : 0) + c * 8;
      int lo = c * 8;
#pragma unroll
      for (int i = 0; i < 8; i++) {
        int g = g0 + i;
        float f = (ev && g < GDIM) ? ap[g] : 0.0f;
        a1[ks][lo + i] = (__bf16)f;
      }
    }
  }

  // --- GEMM1: h = ssp(ea @ W1^T + b1) -> per-wave LDS tile (row-major, bf16)
  __bf16* sh = &s_h[wave][0];
  for (int nt = 0; nt < 8; nt++) {
    int n = nt * 16 + l16;
    float bi = b1[n];
    v8f acc;
#pragma unroll
    for (int i = 0; i < 8; i++) acc[i] = bi;
#pragma unroll
    for (int ks = 0; ks < 2; ks++) {
      const __bf16* wp = w1b + n * GPAD + ks * 32 + (hi ? 16 : 0);
      v16bf bb = *(const v16bf*)wp;  // 32B aligned, L0/L2-resident weights
      acc = __builtin_amdgcn_wmma_f32_16x16x32_bf16(
          false, a1[ks], false, bb, (short)0, acc, false, false);
    }
#pragma unroll
    for (int j = 0; j < 8; j++)
      sh[(j + 8 * hi) * WSTR + n] = (__bf16)sspf(acc[j]);
  }

  // --- transpose C/D layout -> A layout by re-reading the LDS h-tile
  v16bf a2[4];
#pragma unroll
  for (int ks = 0; ks < 4; ks++) {
    int k1 = ks * 32 + (hi ? 16 : 0);
    v8bf c0 = *(const v8bf*)(sh + l16 * WSTR + k1);
    v8bf c1 = *(const v8bf*)(sh + l16 * WSTR + k1 + 8);
    a2[ks] = pack16(c0, c1);
  }

  // --- GEMM2 + cutoff scale + gather(xh[src]) + scatter atomicAdd(agg[dst])
  for (int nt = 0; nt < 8; nt++) {
    int n = nt * 16 + l16;
    float bi = b2[n];
    v8f acc;
#pragma unroll
    for (int i = 0; i < 8; i++) acc[i] = bi;
#pragma unroll
    for (int ks = 0; ks < 4; ks++) {
      int koff = ks * 32 + (hi ? 16 : 0);
      v8bf b0 = *(const v8bf*)(s_w2 + n * WSTR + koff);
      v8bf b1f = *(const v8bf*)(s_w2 + n * WSTR + koff + 8);
      acc = __builtin_amdgcn_wmma_f32_16x16x32_bf16(
          false, a2[ks], false, pack16(b0, b1f), (short)0, acc, false, false);
    }
#pragma unroll
    for (int j = 0; j < 8; j++) {
      float Wv = acc[j] * Cm[j];
      float xv = xh[(long)sm[j] * HDIM + n];   // gather: L2-resident (25.6 MB)
      if (dm[j] >= 0)
        atomicAdd(&agg[(long)dm[j] * HDIM + n], xv * Wv);  // scatter in L2
    }
  }
}

// ---------------------------------------------------------------------------
extern "C" void kernel_launch(void* const* d_in, const int* in_sizes, int n_in,
                              void* d_out, int out_size, void* d_ws, size_t ws_size,
                              hipStream_t stream) {
  const float*     x   = (const float*)d_in[0];
  const long long* ei  = (const long long*)d_in[1];
  const float*     ew  = (const float*)d_in[2];
  const float*     ea  = (const float*)d_in[3];
  const float*     w1  = (const float*)d_in[4];
  const float*     b1  = (const float*)d_in[5];
  const float*     w2  = (const float*)d_in[6];
  const float*     b2  = (const float*)d_in[7];
  const float*     l1  = (const float*)d_in[8];
  const float*     l2  = (const float*)d_in[9];
  const float*     l2b_f = (const float*)d_in[10];
  float* out = (float*)d_out;

  int N = in_sizes[0] / HDIM;
  int E = in_sizes[2];

  // workspace layout
  float*  xh  = (float*)d_ws;
  float*  agg = xh + (size_t)N * HDIM;
  __bf16* w1b = (__bf16*)(agg + (size_t)N * HDIM);
  __bf16* w2b = w1b + HDIM * GPAD;
  __bf16* l1b = w2b + HDIM * HDIM;
  __bf16* l2b = l1b + HDIM * HDIM;

  prep_weights_kernel<<<64, 256, 0, stream>>>(w1, w2, l1, l2, w1b, w2b, l1b, l2b);

  long tot = (long)N * HDIM;
  zero_kernel<<<(int)((tot + 255) / 256), 256, 0, stream>>>(agg, tot);

  int ntiles = (N + 15) / 16;
  gemm_rowtile_kernel<<<(ntiles + 7) / 8, 256, 0, stream>>>(x, l1b, nullptr, xh, N, 0);

  long etiles = ((long)E + 15) / 16;
  edge_kernel<<<(int)((etiles + EWAVES - 1) / EWAVES), EWAVES * 32, 0, stream>>>(
      ea, ew, ei, b1, b2, w1b, w2b, xh, agg, E);

  gemm_rowtile_kernel<<<(ntiles + 7) / 8, 256, 0, stream>>>(agg, l2b, l2b_f, out, N, 1);
}
